// ParallelRiskAggregationNN_22196390986038
// MI455X (gfx1250) — compile-verified, matrix operand order checked by reading, not run on hardware
//
#include <hip/hip_runtime.h>
#include <hip/hip_bf16.h>

typedef __attribute__((ext_vector_type(16))) __bf16 v16bf;
typedef __attribute__((ext_vector_type(8)))  float  v8f;

#define GSTR 1032   // 1024 + 8 bf16 pad (2064B rows -> staggered banks)
#define HSTR 72     // 64 + 8 bf16 pad

union FragBF {
    v16bf v;
    uint4 q[2];
};

// Load a 16x16x32 WMMA A/B fragment for this lane.
// p points at (row_or_col base) + kbase where kbase = kk + (lane>=16 ? 8 : 0).
// halves 0..7  <- K = kbase+0..7   (VGPR0..3)
// halves 8..15 <- K = kbase+16..23 (VGPR4..7)
__device__ __forceinline__ FragBF load_frag(const __bf16* p) {
    FragBF f;
    f.q[0] = *(const uint4*)(p);
    f.q[1] = *(const uint4*)(p + 16);
    return f;
}

// ---------------------------------------------------------------------------
// Weight conversion kernels (f32 -> bf16), run every launch (deterministic).
// ---------------------------------------------------------------------------
__global__ void k_f32_to_bf16(const float* __restrict__ in, __bf16* __restrict__ out, int n) {
    int i = blockIdx.x * 256 + threadIdx.x;
    if (i < n) out[i] = (__bf16)in[i];
}

// gW1 is [1024][16]; pad K to 32 with zeros so layer 1 uses the same WMMA path.
__global__ void k_conv_gW1(const float* __restrict__ in, __bf16* __restrict__ out) {
    int i = blockIdx.x * 256 + threadIdx.x;   // 1024*32 threads
    int r = i >> 5, c = i & 31;
    out[i] = (c < 16) ? (__bf16)in[r * 16 + c] : (__bf16)0.0f;
}

// ---------------------------------------------------------------------------
// g-network: fused 5-layer MLP, 64 rows per workgroup, acts ping-pong in LDS.
// ---------------------------------------------------------------------------
// One layer: act_out[64][1024] = relu(act_in[64][K] * W[1024][K]^T + bias)
__device__ __forceinline__ void g_layer(const __bf16* actIn, __bf16* actOut,
                                        const __bf16* __restrict__ W,
                                        const float*  __restrict__ bias,
                                        int K, int tid)
{
    const int wave = tid >> 5;
    const int lane = tid & 31;
    const int l15  = lane & 15;
    const int koff = (lane >> 4) ? 8 : 0;

    for (int nt = wave; nt < 64; nt += 8) {           // 16-col strip per wave
        const int n = nt * 16 + l15;
        const __bf16* wrow = W + (size_t)n * K;       // row n of W (K bf16)
        v8f acc[4] = {};                              // 4 M-tiles (64 rows)
        for (int kk = 0; kk < K; kk += 32) {
            const int kb = kk + koff;
            FragBF b = load_frag(wrow + kb);          // global (L2-resident)
#pragma unroll
            for (int mt = 0; mt < 4; ++mt) {
                FragBF a = load_frag(actIn + (mt * 16 + l15) * GSTR + kb);
                acc[mt] = __builtin_amdgcn_wmma_f32_16x16x32_bf16(
                    false, a.v, false, b.v, (short)0, acc[mt], false, false);
            }
        }
        const float bv = bias[n];
#pragma unroll
        for (int mt = 0; mt < 4; ++mt)
#pragma unroll
            for (int j = 0; j < 8; ++j) {
                const int m = mt * 16 + j + koff;     // lane>=16 holds rows j+8
                actOut[m * GSTR + n] = (__bf16)fmaxf(acc[mt][j] + bv, 0.0f);
            }
    }
}

__global__ void g_mlp_kernel(const float* __restrict__ x,
                             const __bf16* __restrict__ gW1b, const float* __restrict__ gb1,
                             const __bf16* __restrict__ gW2b, const float* __restrict__ gb2,
                             const __bf16* __restrict__ gW3b, const float* __restrict__ gb3,
                             const __bf16* __restrict__ gW4b, const float* __restrict__ gb4,
                             const __bf16* __restrict__ gW5b, const float* __restrict__ gb5,
                             float* __restrict__ out_g)
{
    extern __shared__ __align__(16) char smem[];
    __bf16* actA = (__bf16*)smem;                 // 64 x GSTR
    __bf16* actB = actA + 64 * GSTR;              // 64 x GSTR

    const int  tid  = threadIdx.x;
    const long row0 = (long)blockIdx.x * 64;

    // Load first half of x (16 f32) as bf16, zero-pad K 16..31.
    for (int i = tid; i < 64 * 32; i += 256) {
        const int r = i >> 5, c = i & 31;
        actA[r * GSTR + c] = (c < 16) ? (__bf16)x[(row0 + r) * 32 + c] : (__bf16)0.0f;
    }
    __syncthreads();
    g_layer(actA, actB, gW1b, gb1,   32, tid); __syncthreads();
    g_layer(actB, actA, gW2b, gb2, 1024, tid); __syncthreads();
    g_layer(actA, actB, gW3b, gb3, 1024, tid); __syncthreads();
    g_layer(actB, actA, gW4b, gb4, 1024, tid); __syncthreads();

    if (tid < 64) {                               // final 1024 -> 1 dot
        float acc = gb5[0];
        const __bf16* ar = actA + tid * GSTR;
        for (int h = 0; h < 1024; ++h) acc += (float)ar[h] * (float)gW5b[h];
        out_g[row0 + tid] = acc;
    }
}

// ---------------------------------------------------------------------------
// h-network: 16 per-dim MLPs 1->64->64->64->64->1, 128 rows per workgroup.
// ---------------------------------------------------------------------------
__device__ __forceinline__ void h_layer(const __bf16* actIn, __bf16* actOut,
                                        const __bf16* __restrict__ W,
                                        const float*  __restrict__ bias, int tid)
{
    const int wave  = tid >> 5;
    const int lane  = tid & 31;
    const int l15   = lane & 15;
    const int koff  = (lane >> 4) ? 8 : 0;
    const int nt    = wave & 3;                   // 4 N-tiles of 16
    const int mbase = (wave >> 2) * 64;           // rows 0..63 or 64..127
    const int n     = nt * 16 + l15;
    const __bf16* wrow = W + n * 64;

    v8f acc[4] = {};
#pragma unroll
    for (int kk = 0; kk < 64; kk += 32) {
        const int kb = kk + koff;
        FragBF b = load_frag(wrow + kb);
#pragma unroll
        for (int mt = 0; mt < 4; ++mt) {
            FragBF a = load_frag(actIn + (mbase + mt * 16 + l15) * HSTR + kb);
            acc[mt] = __builtin_amdgcn_wmma_f32_16x16x32_bf16(
                false, a.v, false, b.v, (short)0, acc[mt], false, false);
        }
    }
    const float bv = bias[n];
#pragma unroll
    for (int mt = 0; mt < 4; ++mt)
#pragma unroll
        for (int j = 0; j < 8; ++j) {
            const int m = mbase + mt * 16 + j + koff;
            actOut[m * HSTR + n] = (__bf16)fmaxf(acc[mt][j] + bv, 0.0f);
        }
}

__global__ void h_mlp_kernel(const float* __restrict__ x,
                             const float*  __restrict__ hW1,  const float* __restrict__ hb1,
                             const __bf16* __restrict__ hW2b, const float* __restrict__ hb2,
                             const __bf16* __restrict__ hW3b, const float* __restrict__ hb3,
                             const __bf16* __restrict__ hW4b, const float* __restrict__ hb4,
                             const float*  __restrict__ hW5,  const float* __restrict__ hb5,
                             float* __restrict__ out_h)
{
    __shared__ __bf16 act1[128 * HSTR];
    __shared__ __bf16 act2[128 * HSTR];

    const int  tid  = threadIdx.x;
    const int  d    = blockIdx.y;
    const long row0 = (long)blockIdx.x * 128;

    const float* w1 = hW1 + d * 64;               // hW1[d][h][0]
    const float* b1 = hb1 + d * 64;

    // Layer 1: scalar input * per-unit weight (VALU), write bf16 to LDS.
    for (int i = tid; i < 128 * 64; i += 256) {
        const int r = i >> 6, c = i & 63;
        const float xv = x[(row0 + r) * 32 + 16 + d];   // second_half[:, d]
        act1[r * HSTR + c] = (__bf16)fmaxf(xv * w1[c] + b1[c], 0.0f);
    }
    __syncthreads();
    h_layer(act1, act2, hW2b + d * 4096, hb2 + d * 64, tid); __syncthreads();
    h_layer(act2, act1, hW3b + d * 4096, hb3 + d * 64, tid); __syncthreads();
    h_layer(act1, act2, hW4b + d * 4096, hb4 + d * 64, tid); __syncthreads();

    if (tid < 128) {                              // final 64 -> 1 dot
        float acc = hb5[d];
        const float*  w5 = hW5 + d * 64;
        const __bf16* ar = act2 + tid * HSTR;
        for (int h = 0; h < 64; ++h) acc += (float)ar[h] * w5[h];
        out_h[(row0 + tid) * 16 + d] = acc;
    }
}

// ---------------------------------------------------------------------------
extern "C" void kernel_launch(void* const* d_in, const int* in_sizes, int n_in,
                              void* d_out, int out_size, void* d_ws, size_t ws_size,
                              hipStream_t stream)
{
    (void)in_sizes; (void)n_in; (void)out_size; (void)ws_size;

    const float* x   = (const float*)d_in[0];
    const float* hW1 = (const float*)d_in[1];
    const float* hb1 = (const float*)d_in[2];
    const float* hW2 = (const float*)d_in[3];
    const float* hb2 = (const float*)d_in[4];
    const float* hW3 = (const float*)d_in[5];
    const float* hb3 = (const float*)d_in[6];
    const float* hW4 = (const float*)d_in[7];
    const float* hb4 = (const float*)d_in[8];
    const float* hW5 = (const float*)d_in[9];
    const float* hb5 = (const float*)d_in[10];
    const float* gW1 = (const float*)d_in[11];
    const float* gb1 = (const float*)d_in[12];
    const float* gW2 = (const float*)d_in[13];
    const float* gb2 = (const float*)d_in[14];
    const float* gW3 = (const float*)d_in[15];
    const float* gb3 = (const float*)d_in[16];
    const float* gW4 = (const float*)d_in[17];
    const float* gb4 = (const float*)d_in[18];
    const float* gW5 = (const float*)d_in[19];
    const float* gb5 = (const float*)d_in[20];

    // Workspace layout (bf16 units, all 16B-aligned offsets): ~6.8 MB total.
    __bf16* ws   = (__bf16*)d_ws;
    __bf16* gW1b = ws;                         // 1024*32 (K padded 16->32)
    __bf16* gW2b = gW1b + 1024 * 32;           // 1024*1024
    __bf16* gW3b = gW2b + 1024 * 1024;
    __bf16* gW4b = gW3b + 1024 * 1024;
    __bf16* gW5b = gW4b + 1024 * 1024;         // 1024
    __bf16* hW2b = gW5b + 1024;                // 16*64*64
    __bf16* hW3b = hW2b + 65536;
    __bf16* hW4b = hW3b + 65536;

    // Convert weights to bf16.
    k_conv_gW1  <<<128,  256, 0, stream>>>(gW1, gW1b);
    k_f32_to_bf16<<<4096, 256, 0, stream>>>(gW2, gW2b, 1024 * 1024);
    k_f32_to_bf16<<<4096, 256, 0, stream>>>(gW3, gW3b, 1024 * 1024);
    k_f32_to_bf16<<<4096, 256, 0, stream>>>(gW4, gW4b, 1024 * 1024);
    k_f32_to_bf16<<<4,    256, 0, stream>>>(gW5, gW5b, 1024);
    k_f32_to_bf16<<<256,  256, 0, stream>>>(hW2, hW2b, 65536);
    k_f32_to_bf16<<<256,  256, 0, stream>>>(hW3, hW3b, 65536);
    k_f32_to_bf16<<<256,  256, 0, stream>>>(hW4, hW4b, 65536);

    float* out_h = (float*)d_out;              // [32768, 16]
    float* out_g = out_h + 32768 * 16;         // [32768]

    const size_t g_shmem = (size_t)2 * 64 * GSTR * sizeof(__bf16); // 264192 B
    (void)hipFuncSetAttribute((const void*)g_mlp_kernel,
                              hipFuncAttributeMaxDynamicSharedMemorySize,
                              (int)g_shmem);

    g_mlp_kernel<<<512, 256, g_shmem, stream>>>(
        x, gW1b, gb1, gW2b, gb2, gW3b, gb3, gW4b, gb4, gW5b, gb5, out_g);

    h_mlp_kernel<<<dim3(256, 16), 256, 0, stream>>>(
        x, hW1, hb1, hW2b, hb2, hW3b, hb3, hW4b, hb4, hW5, hb5, out_h);
}
